// SCARF_baseline_50070728737060
// MI455X (gfx1250) — matrix operand order repaired and checked by reading, not compiled, not run on hardware
//
#include <hip/hip_runtime.h>
#include <hip/hip_bf16.h>

#if __has_include(<hip/amd_detail/amd_gfx1250_TDM.h>)
#define TDM_SIX_ARGS 1
#endif

// ---------------------------------------------------------------------------
// SCARF forward on MI455X (gfx1250, wave32, WMMA).
//   B=4096 rows, K_IN=20000 features, EMB=100 (padded to 112 = 7 x 16 tiles).
// ---------------------------------------------------------------------------
constexpr int BROWS = 4096;
constexpr int KIN   = 20000;
constexpr int EMBD  = 100;
constexpr int EP    = 112;            // EMB padded to 7 WMMA N-tiles
constexpr int KT    = 32;             // bf16 WMMA K depth
constexpr int NSTEP = KIN / KT;       // 625

typedef __attribute__((ext_vector_type(16))) __bf16       v16bf;
typedef __attribute__((ext_vector_type(2)))  __bf16       v2bf;
typedef __attribute__((ext_vector_type(8)))  float        v8f;
typedef __attribute__((ext_vector_type(4)))  unsigned int v4u;
typedef __attribute__((ext_vector_type(8)))  int          v8i;
typedef __attribute__((ext_vector_type(4)))  int          v4i;

__device__ __forceinline__ v8f wmma_bf16(v16bf a, v16bf b, v8f c) {
  return __builtin_amdgcn_wmma_f32_16x16x32_bf16(
      /*neg_a=*/false, a, /*neg_b=*/false, b,
      /*c_mod=*/(short)0, c, /*reuse_a=*/false, /*reuse_b=*/false);
}

// ---------------------------------------------------------------------------
// TDM: DMA a (tile_dim1=112 rows) x (tile_dim0=32 cols) f32 tile of W1
// (row stride 20000 elements, tensor_dim1=100 so rows 100..111 read zero)
// into LDS at lds_addr.  D# packing per CDNA5 ISA ch.8 (group0 + group1,
// groups 2/3 zero for a 2D tensor).
// ---------------------------------------------------------------------------
__device__ __forceinline__ void tdm_load_w_chunk(unsigned lds_addr,
                                                 const float* gsrc) {
  unsigned long long ga = (unsigned long long)(uintptr_t)gsrc;
  v4u g0 = { 1u,                                   // count=1, user desc
             lds_addr,                             // lds_addr [63:32]
             (unsigned)ga,                         // global_addr lo
             (unsigned)((ga >> 32) & 0x01FFFFFFull) | 0x80000000u }; // type=2
  v8i g1 = { (int)(2u << 16),                                  // data_size=4B
             (int)((unsigned)(KIN & 0xFFFF) << 16),            // tensor_dim0 lo16
             (int)(((unsigned)KIN >> 16) | ((unsigned)EMBD << 16)), // dim0 hi | dim1 lo
             (int)((unsigned)KT << 16),                        // tile_dim0=32
             (int)(unsigned)EP,                                // tile_dim1=112
             (int)(unsigned)KIN,                               // dim0 stride lo32
             0, 0 };
  v4i z4 = {0, 0, 0, 0};
#if defined(TDM_SIX_ARGS)
  v8i z8 = {0, 0, 0, 0, 0, 0, 0, 0};
  __builtin_amdgcn_tensor_load_to_lds(g0, g1, z4, z4, z8, 0);
#else
  __builtin_amdgcn_tensor_load_to_lds(g0, g1, z4, z4, 0);
#endif
}

// ---------------------------------------------------------------------------
// Kernel 1: A1[p][row][e] = relu( Xp @ W1^T + b1 ),  p=0 clean, p=1 corrupted.
// One pass over x / x_random / mask computes BOTH paths (shared B-fragments).
// Pipeline per 32-K chunk:
//   TDM raw f32 -> LDS  |  whole block converts once to bf16 frag layout in
//   LDS (no redundant per-wave cvt)  |  waves consume via 2x ds_load_b128.
// Grid: 32 blocks x 256 threads (8 waves); wave w owns rows [blk*128+w*16, +16).
// ---------------------------------------------------------------------------
__global__ __launch_bounds__(256) void k_gemm1(
    const float* __restrict__ x, const unsigned char* __restrict__ mask,
    const float* __restrict__ xr, const float* __restrict__ w1,
    const float* __restrict__ b1, float* __restrict__ A1) {
  __shared__ float  shr[2][EP * KT];        // raw f32 (TDM dest), 2 x 14336 B
  __shared__ __bf16 shf[2][7 * 32 * 16];    // bf16 fragments,     2 x  7168 B

  const int tid  = threadIdx.x;
  const int wave = tid >> 5;
  const int lane = tid & 31;
  const int r0   = blockIdx.x * 128 + wave * 16;
  const int mrow = r0 + (lane & 15);
  const int koff = (lane >> 4) << 3;             // A-frag K offset: 0 or 8

  v8f acc0[7], acc1[7];
#pragma unroll
  for (int t = 0; t < 7; ++t) {
    acc0[t] = v8f{0.f, 0.f, 0.f, 0.f, 0.f, 0.f, 0.f, 0.f};
    acc1[t] = acc0[t];
  }

  const unsigned ldsBase = (unsigned)(uintptr_t)(&shr[0][0]);
  if (wave == 0) tdm_load_w_chunk(ldsBase, w1);  // prologue: chunk 0

  const size_t rowBase = (size_t)mrow * KIN;
  for (int i = 0; i < NSTEP; ++i) {
    if (wave == 0) __builtin_amdgcn_s_wait_tensorcnt(0);   // raw[i&1] landed
    __syncthreads();
    if (wave == 0 && (i + 1) < NSTEP)                      // overlap DMA i+1
      tdm_load_w_chunk(ldsBase + (unsigned)(((i + 1) & 1) * (EP * KT * 4)),
                       w1 + (size_t)(i + 1) * KT);
    const int buf = i & 1;
    const int k0  = i * KT;

    // ---- block-wide convert: raw f32 -> bf16 fragment layout (once) ------
#pragma unroll
    for (int q = tid; q < (EP * KT / 2); q += 256) {       // 1792 f32 pairs
      const int e  = q >> 4;                               // W row 0..111
      const int kp = (q & 15) << 1;                        // even k 0..30
      const float f0 = shr[buf][e * KT + kp];
      const float f1 = shr[buf][e * KT + kp + 1];
      v2bf pk;
      pk[0] = (__bf16)f0;
      pk[1] = (__bf16)f1;
      const int t = e >> 4;
      const int l = (e & 15) + (((kp >> 3) & 1) ? 16 : 0); // dest lane
      const int h = (kp & 7) + ((kp >= 16) ? 8 : 0);       // half index
      *(v2bf*)&shf[buf][(t * 32 + l) * 16 + h] = pk;
    }
    __syncthreads();                                       // frag[buf] ready

    // ---- A fragments: clean + on-the-fly corrupted (where(mask,xr,x)) ----
    const size_t ab = rowBase + k0 + koff;
    float xv[16], rv[16];
    unsigned char mb[16];
    *(float4*)&xv[0]  = *(const float4*)(x + ab);
    *(float4*)&xv[4]  = *(const float4*)(x + ab + 4);
    *(float4*)&xv[8]  = *(const float4*)(x + ab + 16);
    *(float4*)&xv[12] = *(const float4*)(x + ab + 20);
    *(float4*)&rv[0]  = *(const float4*)(xr + ab);
    *(float4*)&rv[4]  = *(const float4*)(xr + ab + 4);
    *(float4*)&rv[8]  = *(const float4*)(xr + ab + 16);
    *(float4*)&rv[12] = *(const float4*)(xr + ab + 20);
    *(uint2*)&mb[0]   = *(const uint2*)(mask + ab);
    *(uint2*)&mb[8]   = *(const uint2*)(mask + ab + 16);
    v16bf fa, fc;
#pragma unroll
    for (int e = 0; e < 16; ++e) {
      fa[e] = (__bf16)xv[e];
      fc[e] = (__bf16)(mb[e] ? rv[e] : xv[e]);
    }

    // ---- consume fragments: pure ds_load_b128 + 14 WMMAs -----------------
#pragma unroll
    for (int t = 0; t < 7; ++t) {
      const v16bf fb = *(const v16bf*)&shf[buf][(t * 32 + lane) * 16];
      acc0[t] = wmma_bf16(fa, fb, acc0[t]);
      acc1[t] = wmma_bf16(fc, fb, acc1[t]);
    }
  }

  // ---- epilogue: + bias, ReLU, store both paths (pad cols store exact 0) --
#pragma unroll
  for (int t = 0; t < 7; ++t) {
    const int col = t * 16 + (lane & 15);
    const float bb = (col < EMBD) ? b1[col] : 0.f;
#pragma unroll
    for (int j = 0; j < 8; ++j) {
      const int orow = r0 + ((lane >> 4) << 3) + j;
      float v0 = fmaxf(acc0[t][j] + bb, 0.f);
      float v1 = fmaxf(acc1[t][j] + bb, 0.f);
      A1[(size_t)orow * EP + col] = v0;
      A1[(size_t)(BROWS + orow) * EP + col] = v1;
    }
  }
}

// ---------------------------------------------------------------------------
// BN stats: per (path, feature) biased mean/var over 4096 rows ->
//   scale = gamma * rsqrt(var+eps), shift = beta - mean*scale.
// Pad features (col>=100) get scale=shift=0 so padded K lanes contribute 0.
// ---------------------------------------------------------------------------
__global__ __launch_bounds__(256) void k_stats(
    const float* __restrict__ A, const float* __restrict__ gamma,
    const float* __restrict__ beta, float* __restrict__ stats) {
  const int col = blockIdx.x;
  const int p   = blockIdx.y;
  const int tid = threadIdx.x;
  const float* Ap = A + (size_t)p * BROWS * EP + col;
  float s = 0.f, ss = 0.f;
  for (int r = tid; r < BROWS; r += 256) {
    float v = Ap[(size_t)r * EP];
    s += v;
    ss += v * v;
  }
  __shared__ float sh1[256], sh2[256];
  sh1[tid] = s;
  sh2[tid] = ss;
  __syncthreads();
  for (int o = 128; o > 0; o >>= 1) {
    if (tid < o) {
      sh1[tid] += sh1[tid + o];
      sh2[tid] += sh2[tid + o];
    }
    __syncthreads();
  }
  if (tid == 0) {
    const float inv  = 1.f / (float)BROWS;
    const float mean = sh1[0] * inv;
    const float var  = sh2[0] * inv - mean * mean;
    float sc = 0.f, sf = 0.f;
    if (col < EMBD) {
      const float r = rsqrtf(var + 1e-5f);
      sc = gamma[col] * r;
      sf = beta[col] - mean * sc;
    }
    float* st = stats + (size_t)p * 2 * EP;
    st[col]      = sc;
    st[EP + col] = sf;
  }
}

// ---------------------------------------------------------------------------
// Small GEMM: out = f_post( BN_pre(A) @ W^T + bias ),  K=100 padded to 128.
// The full padded W (112 out-cols x 128 K) is staged ONCE per block into LDS
// in bf16 fragment order; the K-loop is then ds_load_b128 + WMMA only.
//   preRelu : ReLU after BN on the A side (head block order BN->ReLU->GEMM)
//   postRelu: ReLU on the output (encoder blocks: GEMM->ReLU, BN later)
//   guardOut: only store cols < 100 (final d_out, ldOut=100)
// ---------------------------------------------------------------------------
__global__ __launch_bounds__(256) void k_gemm_small(
    const float* __restrict__ A, const float* __restrict__ stats,
    const int preRelu, const float* __restrict__ W,
    const float* __restrict__ bias, float* __restrict__ out, const int ldOut,
    const int postRelu, const int guardOut) {
  __shared__ __bf16 wf[4 * 7 * 32 * 16];          // 28672 B, all 4 K-steps

  const int p = blockIdx.y;
  const float* Ap    = A + (size_t)p * BROWS * EP;
  const float* scale = stats + (size_t)p * 2 * EP;
  const float* shift = scale + EP;
  const int tid  = threadIdx.x;
  const int wave = tid >> 5;
  const int lane = tid & 31;
  const int r0   = blockIdx.x * 128 + wave * 16;
  const int mrow = r0 + (lane & 15);
  const int koff = (lane >> 4) << 3;

  // ---- stage padded bf16 W fragments once per block ----------------------
#pragma unroll
  for (int q = tid; q < 4 * 7 * 32 * 8; q += 256) {   // 7168 f32 pairs
    const int h  = (q & 7) << 1;                      // half index 0..14
    int r        = q >> 3;
    const int l  = r & 31;                            // dest lane
    r >>= 5;
    const int t  = r % 7;                             // N tile
    const int ks = r / 7;                             // K step
    const int ecol = t * 16 + (l & 15);
    const int k    = ks * KT + ((l >= 16) ? 8 : 0) + ((h >= 8) ? 8 : 0) + h;
    const bool eok = (ecol < EMBD);
    const float w0 = (eok && k < EMBD)       ? W[(size_t)ecol * EMBD + k]     : 0.f;
    const float w1 = (eok && (k + 1) < EMBD) ? W[(size_t)ecol * EMBD + k + 1] : 0.f;
    v2bf pk;
    pk[0] = (__bf16)w0;
    pk[1] = (__bf16)w1;
    *(v2bf*)&wf[((ks * 7 + t) * 32 + l) * 16 + h] = pk;
  }
  __syncthreads();

  v8f acc[7];
#pragma unroll
  for (int t = 0; t < 7; ++t)
    acc[t] = v8f{0.f, 0.f, 0.f, 0.f, 0.f, 0.f, 0.f, 0.f};

#pragma unroll
  for (int ks = 0; ks < 4; ++ks) {
    const int k0 = ks * KT;
    v16bf fa;
#pragma unroll
    for (int e = 0; e < 16; ++e) {
      const int k = k0 + koff + ((e < 8) ? e : (e + 8));
      float v = 0.f;
      if (k < EP) {
        v = Ap[(size_t)mrow * EP + k] * scale[k] + shift[k];
        if (preRelu) v = fmaxf(v, 0.f);
      }
      fa[e] = (__bf16)v;
    }
#pragma unroll
    for (int t = 0; t < 7; ++t) {
      const v16bf fb = *(const v16bf*)&wf[((ks * 7 + t) * 32 + lane) * 16];
      acc[t] = wmma_bf16(fa, fb, acc[t]);
    }
  }

  float* op = out + (size_t)p * BROWS * ldOut;
#pragma unroll
  for (int t = 0; t < 7; ++t) {
    const int col = t * 16 + (lane & 15);
    if (guardOut && col >= EMBD) continue;
    const float bb = (col < EMBD) ? bias[col] : 0.f;
#pragma unroll
    for (int j = 0; j < 8; ++j) {
      const int orow = r0 + ((lane >> 4) << 3) + j;
      float v = acc[t][j] + bb;
      if (postRelu) v = fmaxf(v, 0.f);
      op[(size_t)orow * ldOut + col] = v;
    }
  }
}

// ---------------------------------------------------------------------------
extern "C" void kernel_launch(void* const* d_in, const int* in_sizes, int n_in,
                              void* d_out, int out_size, void* d_ws,
                              size_t ws_size, hipStream_t stream) {
  const float*         x    = (const float*)d_in[0];
  const unsigned char* mask = (const unsigned char*)d_in[1];   // jnp.bool_
  const float*         xr   = (const float*)d_in[2];
  const float* ew1 = (const float*)d_in[3];
  const float* eb1 = (const float*)d_in[4];
  const float* eg1 = (const float*)d_in[5];
  const float* ebe1 = (const float*)d_in[6];
  const float* ew2 = (const float*)d_in[7];
  const float* eb2 = (const float*)d_in[8];
  const float* eg2 = (const float*)d_in[9];
  const float* ebe2 = (const float*)d_in[10];
  const float* hw1 = (const float*)d_in[11];
  const float* hb1 = (const float*)d_in[12];
  const float* hg1 = (const float*)d_in[13];
  const float* hbe1 = (const float*)d_in[14];
  const float* hw2 = (const float*)d_in[15];
  const float* hb2 = (const float*)d_in[16];

  float* ws = (float*)d_ws;
  const size_t matElems = (size_t)2 * BROWS * EP;   // 917504 floats
  float* A1  = ws;
  float* A2  = A1 + matElems;
  float* G1  = A2 + matElems;
  float* st0 = G1 + matElems;                       // each stats = 2*2*112
  float* st1 = st0 + 2 * 2 * EP;
  float* st2 = st1 + 2 * 2 * EP;

  // 1) big fused GEMM: both paths, TDM-staged W1, one pass over x/xr/mask
  k_gemm1<<<dim3(BROWS / 128), dim3(256), 0, stream>>>(x, mask, xr, ew1, eb1,
                                                       A1);
  // 2) encoder BN1 stats
  k_stats<<<dim3(EP, 2), dim3(256), 0, stream>>>(A1, eg1, ebe1, st0);
  // 3) A2 = relu( BN1(A1) @ W2^T + b2 )
  k_gemm_small<<<dim3(BROWS / 128, 2), dim3(256), 0, stream>>>(
      A1, st0, 0, ew2, eb2, A2, EP, 1, 0);
  // 4) encoder BN2 stats
  k_stats<<<dim3(EP, 2), dim3(256), 0, stream>>>(A2, eg2, ebe2, st1);
  // 5) G1 = BN2(A2) @ HW1^T + hb1
  k_gemm_small<<<dim3(BROWS / 128, 2), dim3(256), 0, stream>>>(
      A2, st1, 0, hw1, hb1, G1, EP, 0, 0);
  // 6) head BN stats
  k_stats<<<dim3(EP, 2), dim3(256), 0, stream>>>(G1, hg1, hbe1, st2);
  // 7) out = relu(BN3(G1)) @ HW2^T + hb2  -> d_out (clean then corrupted)
  k_gemm_small<<<dim3(BROWS / 128, 2), dim3(256), 0, stream>>>(
      G1, st2, 1, hw2, hb2, (float*)d_out, EMBD, 0, 1);
}
